// SharedScaleNA_87720412054034
// MI455X (gfx1250) — matrix operand order repaired and checked by reading, not compiled, not run on hardware
//
#include <hip/hip_runtime.h>
#include <hip/hip_bf16.h>

typedef __attribute__((ext_vector_type(16))) _Float16 v16h;
typedef __attribute__((ext_vector_type(8)))  _Float16 v8h;
typedef __attribute__((ext_vector_type(8)))  float    v8f;

#define BN    2
#define HH    48
#define WW    48
#define NPIX  (BN*HH*WW)      // 4608 tokens
#define CMID  256
#define CQKV  768
#define HEADS 8
#define HD    32
#define KWIN  7

// ---------------------------------------------------------------------
// CDNA5 f16 WMMA fragment layouts (CDNA5 ISA §7.12.2), wave32:
//  A (16x32): per lane, elements 0..7  = K [8g, 8g+8),
//                      elements 8..15 = K [16+8g, 16+8g+8)   (g = lane>=16)
//  B (32x16): per lane, same K split as A, N = lane&15
//  C/D (16x16 f32): element r -> M = r + 8g, N = lane&15
// => fragments are contiguous 16B runs; load them with b128s.
// ---------------------------------------------------------------------
__device__ __forceinline__ v16h cat8(v8h lo, v8h hi) {
  v16h r;
  #pragma unroll
  for (int i = 0; i < 8; ++i) { r[i] = lo[i]; r[8 + i] = hi[i]; }
  return r;
}
__device__ __forceinline__ v16h ldA16(const _Float16* rowp, int g) {
  return cat8(*(const v8h*)(rowp + g * 8), *(const v8h*)(rowp + 16 + g * 8));
}
__device__ __forceinline__ v16h ldB16(const _Float16* colp, int g) {
  return cat8(*(const v8h*)(colp + g * 8), *(const v8h*)(colp + 16 + g * 8));
}
__device__ __forceinline__ void fill8(v16h& a, int base, const float* r0, const float* r1) {
  const float4 f0 = *(const float4*)r0;
  const float4 f1 = *(const float4*)r1;
  a[base + 0] = (_Float16)f0.x; a[base + 1] = (_Float16)f0.y;
  a[base + 2] = (_Float16)f0.z; a[base + 3] = (_Float16)f0.w;
  a[base + 4] = (_Float16)f1.x; a[base + 5] = (_Float16)f1.y;
  a[base + 6] = (_Float16)f1.z; a[base + 7] = (_Float16)f1.w;
}
__device__ __forceinline__ v8f v8f_zero() {
  v8f z = {0.f, 0.f, 0.f, 0.f, 0.f, 0.f, 0.f, 0.f};
  return z;
}
__device__ __forceinline__ v8h v8h_zero() {
  v8h z = {(_Float16)0.f, (_Float16)0.f, (_Float16)0.f, (_Float16)0.f,
           (_Float16)0.f, (_Float16)0.f, (_Float16)0.f, (_Float16)0.f};
  return z;
}

// =====================================================================
// Kernel 0: one-shot weight conversion to f16.
//  w_tok (256x48 f32) -> wtH [256][64] f16, zero-padded K 48..63
//  w_qkv (768x256 f32) -> wqH [768][256] f16
// =====================================================================
__global__ void __launch_bounds__(256) na_prep_kernel(
    const float* __restrict__ wt, const float* __restrict__ wq,
    _Float16* __restrict__ wtH, _Float16* __restrict__ wqH) {
  const int stride = gridDim.x * 256;
  for (int i = blockIdx.x * 256 + threadIdx.x; i < 256 * 64; i += stride) {
    const int n = i >> 6, k = i & 63;
    wtH[i] = (k < 48) ? (_Float16)wt[n * 48 + k] : (_Float16)0.f;
  }
  for (int i = blockIdx.x * 256 + threadIdx.x; i < 768 * 256; i += stride) {
    wqH[i] = (_Float16)wq[i];
  }
}

// =====================================================================
// Kernel 1: patch-embed conv as GEMM  tok[4608][256] = im2col(x)[4608][48] * Wt + b
// im2col fused into A-fragment: each 8-elem chunk = two float4 rows of x.
// K pad (48..63) is compile-time zeros. 2 WMMAs / wave.
// =====================================================================
__global__ void __launch_bounds__(256) na_tok_kernel(
    const float* __restrict__ x, const _Float16* __restrict__ wtH,
    const float* __restrict__ bt, _Float16* __restrict__ tok) {
  const int lane = threadIdx.x & 31;
  const int wave = blockIdx.x * 8 + (threadIdx.x >> 5);   // 0..4607
  const int mt = wave >> 4;                               // 288 M-tiles
  const int nt = wave & 15;                               // 16 N-tiles
  const int row = lane & 15;
  const int g   = (lane >> 4) & 1;
  const int m = mt * 16 + row;
  const int n = nt * 16 + row;
  const int bb  = m / (HH * WW);
  const int rem = m % (HH * WW);
  const int hy = rem / WW, wx = rem % WW;

  const float* xb = x + (size_t)bb * 3 * 192 * 192;
  #define XROW(c, py) (xb + ((size_t)(c) * 192 + (hy * 4 + (py))) * 192 + wx * 4)

  v8f acc = v8f_zero();
  // kt = 0 : K 0..31  (c=0 chunk, c=1 chunk)
  {
    v16h a, bf;
    fill8(a, 0, XROW(0, g * 2), XROW(0, g * 2 + 1));
    fill8(a, 8, XROW(1, g * 2), XROW(1, g * 2 + 1));
    bf = ldB16(wtH + n * 64, g);
    acc = __builtin_amdgcn_wmma_f32_16x16x32_f16(false, a, false, bf,
                                                 (short)0, acc, false, false);
  }
  // kt = 1 : K 32..63 (c=2 chunk, zero pad chunk)
  {
    v16h a, bf;
    fill8(a, 0, XROW(2, g * 2), XROW(2, g * 2 + 1));
    #pragma unroll
    for (int i = 8; i < 16; ++i) a[i] = (_Float16)0.f;
    bf = ldB16(wtH + n * 64 + 32, g);
    acc = __builtin_amdgcn_wmma_f32_16x16x32_f16(false, a, false, bf,
                                                 (short)0, acc, false, false);
  }
  #undef XROW
  const float bias = bt[n];
  #pragma unroll
  for (int r = 0; r < 8; ++r) {
    const int mm = mt * 16 + r + (g << 3);
    tok[(size_t)mm * CMID + n] = (_Float16)(acc[r] + bias);
  }
}

// =====================================================================
// Kernel 2: 1x1 qkv conv as GEMM  qkv[4608][768] = tok[4608][256] * Wq + b
// 86% of total FLOPs -> register-blocked 32x32 tile per wave:
// per K-step 2 A-frags + 2 B-frags feed 4 WMMAs (2x fragment reuse).
// 32 WMMAs / wave, all-f16 b128 operand streams.
// =====================================================================
__global__ void __launch_bounds__(256) na_qkv_kernel(
    const _Float16* __restrict__ tok, const _Float16* __restrict__ wqH,
    const float* __restrict__ bq, _Float16* __restrict__ qkv) {
  const int lane = threadIdx.x & 31;
  const int wave = blockIdx.x * 8 + (threadIdx.x >> 5);   // 0..3455
  const int mt = wave / 24;                               // 144 M-tiles of 32
  const int nt = wave % 24;                               // 24 N-tiles of 32
  const int row = lane & 15;
  const int g   = (lane >> 4) & 1;
  const int m0 = mt * 32 + row;
  const int n0 = nt * 32 + row;
  const _Float16* arow0 = tok + (size_t)m0 * CMID;
  const _Float16* arow1 = arow0 + 16 * CMID;
  const _Float16* bcol0 = wqH + (size_t)n0 * CMID;
  const _Float16* bcol1 = bcol0 + 16 * CMID;

  v8f acc00 = v8f_zero(), acc01 = v8f_zero();
  v8f acc10 = v8f_zero(), acc11 = v8f_zero();
  #pragma unroll
  for (int kt = 0; kt < 8; ++kt) {
    const v16h a0 = ldA16(arow0 + kt * 32, g);
    const v16h a1 = ldA16(arow1 + kt * 32, g);
    const v16h b0 = ldB16(bcol0 + kt * 32, g);
    const v16h b1 = ldB16(bcol1 + kt * 32, g);
    acc00 = __builtin_amdgcn_wmma_f32_16x16x32_f16(false, a0, false, b0, (short)0, acc00, false, false);
    acc01 = __builtin_amdgcn_wmma_f32_16x16x32_f16(false, a0, false, b1, (short)0, acc01, false, false);
    acc10 = __builtin_amdgcn_wmma_f32_16x16x32_f16(false, a1, false, b0, (short)0, acc10, false, false);
    acc11 = __builtin_amdgcn_wmma_f32_16x16x32_f16(false, a1, false, b1, (short)0, acc11, false, false);
  }
  const float bias0 = bq[n0];
  const float bias1 = bq[n0 + 16];
  #pragma unroll
  for (int r = 0; r < 8; ++r) {
    const int mmA = mt * 32 + r + (g << 3);
    const int mmB = mmA + 16;
    qkv[(size_t)mmA * CQKV + n0]      = (_Float16)(acc00[r] + bias0);
    qkv[(size_t)mmA * CQKV + n0 + 16] = (_Float16)(acc01[r] + bias1);
    qkv[(size_t)mmB * CQKV + n0]      = (_Float16)(acc10[r] + bias0);
    qkv[(size_t)mmB * CQKV + n0 + 16] = (_Float16)(acc11[r] + bias1);
  }
}

// =====================================================================
// Kernel 3: 7x7 neighborhood attention. One wave per (b, head, 4x4 query tile).
// 16 queries' clipped windows union into <=10x10=100 keys (pad 112/128).
// scores (7 WMMA) -> masked softmax (shfl_xor in 16-lane halves) -> out (8 WMMA).
// All guards are clamp-address + cndmask-value (branch-free).
// =====================================================================
__global__ void __launch_bounds__(256) na_attn_kernel(
    const _Float16* __restrict__ qkv, float* __restrict__ out) {
  __shared__ _Float16 attnLds[8][16 * 128];               // 32 KB, per-wave slices
  const int lane = threadIdx.x & 31;
  const int wsl  = threadIdx.x >> 5;
  const int wave = blockIdx.x * 8 + wsl;                  // 0..2303
  const int tx = wave % 12;
  const int t1 = wave / 12;
  const int ty = t1 % 12;
  const int t2 = t1 / 12;
  const int hh = t2 & 7;
  const int bb = t2 >> 3;

  const int rowbase = min(max(ty * 4 - 3, 0), HH - KWIN);
  const int colbase = min(max(tx * 4 - 3, 0), WW - KWIN);
  const int col = lane & 15;
  const int g   = (lane >> 4) & 1;
  const size_t bpix = (size_t)bb * (HH * WW);

  // per-lane union-key geometry for the 7 score tiles (hoisted once)
  int kyT[7], kxT[7]; bool inT[7];
  #pragma unroll
  for (int t = 0; t < 7; ++t) {
    const int u = t * 16 + col;
    const int uy = u / 10, ux = u % 10;
    kyT[t] = rowbase + uy;
    kxT[t] = colbase + ux;
    inT[t] = (u < 100) && (kyT[t] < HH) && (kxT[t] < WW);
  }

  // ---- GEMM1: scores[16q][112u] = Q(16x32) x K^T(32x112) ----
  v16h aq;
  {
    const int qy = ty * 4 + (col >> 2), qx = tx * 4 + (col & 3);
    aq = ldA16(qkv + (bpix + qy * WW + qx) * CQKV + hh * HD, g);
  }
  v8f sacc[7];
  #pragma unroll
  for (int t = 0; t < 7; ++t) {
    const size_t kpix = bpix + min(kyT[t], HH - 1) * WW + min(kxT[t], WW - 1);
    const _Float16* kp = qkv + kpix * CQKV + CMID + hh * HD;
    v8h lo = *(const v8h*)(kp + g * 8);
    v8h hi = *(const v8h*)(kp + 16 + g * 8);
    lo = inT[t] ? lo : v8h_zero();
    hi = inT[t] ? hi : v8h_zero();
    sacc[t] = __builtin_amdgcn_wmma_f32_16x16x32_f16(false, aq, false, cat8(lo, hi),
                                                     (short)0, v8f_zero(), false, false);
  }

  // ---- masked softmax; C-row M = r + 8g lives across lanes [16g,16g+16) ----
  const float scale = 0.17677669529663687f;               // 1/sqrt(32)
  #pragma unroll
  for (int r = 0; r < 8; ++r) {
    const int mq = r + (g << 3);
    const int qy = ty * 4 + (mq >> 2), qx = tx * 4 + (mq & 3);
    const int sy = min(max(qy - 3, 0), HH - KWIN);
    const int sx = min(max(qx - 3, 0), WW - KWIN);
    float v[7]; bool val[7];
    float mx = -3.0e38f;
    #pragma unroll
    for (int t = 0; t < 7; ++t) {
      val[t] = inT[t] && (kyT[t] >= sy) && (kyT[t] < sy + KWIN)
                      && (kxT[t] >= sx) && (kxT[t] < sx + KWIN);
      v[t] = val[t] ? sacc[t][r] * scale : -3.0e38f;
      mx = fmaxf(mx, v[t]);
    }
    #pragma unroll
    for (int off = 1; off < 16; off <<= 1)                // stays inside 16-lane half
      mx = fmaxf(mx, __shfl_xor(mx, off, 32));
    float sum = 0.f, ex[7];
    #pragma unroll
    for (int t = 0; t < 7; ++t) {
      ex[t] = val[t] ? __expf(v[t] - mx) : 0.f;
      sum += ex[t];
    }
    #pragma unroll
    for (int off = 1; off < 16; off <<= 1)
      sum += __shfl_xor(sum, off, 32);
    const float inv = 1.0f / sum;
    #pragma unroll
    for (int t = 0; t < 7; ++t)
      attnLds[wsl][mq * 128 + t * 16 + col] = (_Float16)(ex[t] * inv);
    attnLds[wsl][mq * 128 + 112 + col] = (_Float16)0.f;   // zero K-pad 112..127
  }
  __asm volatile("s_wait_dscnt 0x0" ::: "memory");        // LDS writes -> fragment reads

  // ---- GEMM2: out[16q][32d] = attn(16x128, f16 LDS) x V(128x32) ----
  v8f oacc[2];
  oacc[0] = v8f_zero();
  oacc[1] = v8f_zero();
  #pragma unroll
  for (int kt = 0; kt < 4; ++kt) {
    const _Float16* ar = &attnLds[wsl][col * 128 + kt * 32];
    v16h aa = cat8(*(const v8h*)(ar + g * 16), *(const v8h*)(ar + g * 16 + 8));
    #pragma unroll
    for (int t = 0; t < 2; ++t) {
      v16h bv;
      #pragma unroll
      for (int e = 0; e < 16; ++e) {
        const int u = kt * 32 + (g << 4) + e;             // union key for this element
        const int uy = u / 10, ux = u % 10;
        const int ky = rowbase + uy, kx = colbase + ux;
        const bool kval = (u < 100) && (ky < HH) && (kx < WW);
        const size_t kpix = bpix + min(ky, HH - 1) * WW + min(kx, WW - 1);
        const _Float16 vv = qkv[kpix * CQKV + 2 * CMID + hh * HD + t * 16 + col];
        bv[e] = kval ? vv : (_Float16)0.f;
      }
      oacc[t] = __builtin_amdgcn_wmma_f32_16x16x32_f16(false, aa, false, bv,
                                                       (short)0, oacc[t], false, false);
    }
  }

  // ---- store out: layout (HEADS, B, HD, H, W) ----
  #pragma unroll
  for (int t = 0; t < 2; ++t) {
    #pragma unroll
    for (int r = 0; r < 8; ++r) {
      const int mq = r + (g << 3);
      const int qy = ty * 4 + (mq >> 2), qx = tx * 4 + (mq & 3);
      const int d = t * 16 + col;
      out[((size_t)(hh * BN + bb) * HD + d) * (HH * WW) + qy * WW + qx] = oacc[t][r];
    }
  }
}

extern "C" void kernel_launch(void* const* d_in, const int* in_sizes, int n_in,
                              void* d_out, int out_size, void* d_ws, size_t ws_size,
                              hipStream_t stream) {
  (void)in_sizes; (void)n_in; (void)out_size; (void)ws_size;
  const float* x     = (const float*)d_in[0];
  const float* w_tok = (const float*)d_in[1];
  const float* b_tok = (const float*)d_in[2];
  const float* w_qkv = (const float*)d_in[3];
  const float* b_qkv = (const float*)d_in[4];
  float* out = (float*)d_out;

  // ws layout (f16 units): tok[4608*256] | qkv[4608*768] | wtH[256*64] | wqH[768*256]
  _Float16* tok = (_Float16*)d_ws;
  _Float16* qkv = tok + (size_t)NPIX * CMID;
  _Float16* wtH = qkv + (size_t)NPIX * CQKV;
  _Float16* wqH = wtH + 256 * 64;

  na_prep_kernel<<<96, 256, 0, stream>>>(w_tok, w_qkv, wtH, wqH);
  // K1: 288 M-tiles * 16 N-tiles = 4608 waves -> 576 blocks of 8 waves
  na_tok_kernel<<<576, 256, 0, stream>>>(x, wtH, b_tok, tok);
  // K2: 144 * 24 = 3456 waves (32x32 tiles) -> 432 blocks
  na_qkv_kernel<<<432, 256, 0, stream>>>(tok, wqH, b_qkv, qkv);
  // K3: 2 * 8 * 12 * 12 = 2304 waves -> 288 blocks
  na_attn_kernel<<<288, 256, 0, stream>>>(qkv, out);
}
